// GRUDecoder_29506425324036
// MI455X (gfx1250) — matrix operand order repaired and checked by reading, not compiled, not run on hardware
//
#include <hip/hip_runtime.h>
#include <hip/hip_bf16.h>
#include <math.h>

// ---------------------------------------------------------------------------
// GRU decoder, B=1024, T=25, IN=OUT=96, H=2048, bf16 WMMA (f32 accum)
// ---------------------------------------------------------------------------

typedef __attribute__((ext_vector_type(8)))  __bf16 v8bf;
typedef __attribute__((ext_vector_type(16))) __bf16 v16bf;
typedef __attribute__((ext_vector_type(8)))  float  v8f;

#define B_    1024
#define T_    25
#define IN_   96
#define OUT_  96
#define HH    2048
#define H3    (3 * HH)

// ---- helpers ---------------------------------------------------------------

__device__ __forceinline__ __bf16 f2bf(float f) {
  // round-to-nearest-even fp32 -> bf16
  unsigned int u = __float_as_uint(f);
  unsigned int r = ((u >> 16) & 1u) + 0x7FFFu;
  unsigned short s = (unsigned short)((u + r) >> 16);
  return __builtin_bit_cast(__bf16, s);
}

__device__ __forceinline__ v16bf make16(v8bf lo, v8bf hi) {
  return __builtin_shufflevector(lo, hi, 0, 1, 2, 3, 4, 5, 6, 7,
                                         8, 9, 10, 11, 12, 13, 14, 15);
}

__device__ __forceinline__ float sigmoidf_(float x) {
  return 1.0f / (1.0f + expf(-x));
}

// ---- fp32 -> bf16 convert --------------------------------------------------

__global__ void k_cvt_bf16(const float* __restrict__ s, __bf16* __restrict__ d, int n) {
  int i = blockIdx.x * blockDim.x + threadIdx.x;
  if (i < n) d[i] = f2bf(s[i]);
}

// hiddens [2, B, H] -> working fp32 copies + bf16 copies
__global__ void k_init_h(const float* __restrict__ hid,
                         float* __restrict__ h0f, float* __restrict__ h1f,
                         __bf16* __restrict__ h0b, __bf16* __restrict__ h1b) {
  int i = blockIdx.x * blockDim.x + threadIdx.x;   // < B_*HH
  float a = hid[i];
  float b = hid[B_ * HH + i];
  h0f[i] = a; h1f[i] = b;
  h0b[i] = f2bf(a); h1b[i] = f2bf(b);
}

// ---- main GEMM: C[M,6144] = A[M,K](bf16) * W[6144,K]^T(bf16) + bias --------
// Tile: 128(M) x 128(N), K-step 32.  8 waves, each wave: 4(M) x 2(N) WMMA tiles.
// K is a template parameter (96 or 2048) => pure shift/add addressing.
// Register-prefetch + LDS double-buffer: next tile's global loads overlap the
// current tile's WMMAs; one workgroup barrier per K-step.
// LDS rows padded to 40 bf16 (80 B) => conflict-free ds_load_b128 frag reads.

#define LDSW 40
#define LBUF (128 * LDSW)   // elements per buffer

template <int K>
__global__ __launch_bounds__(256)
void k_gemm(const __bf16* __restrict__ A, const __bf16* __restrict__ W,
            const float* __restrict__ bias, float* __restrict__ C) {
  constexpr int N = H3;
  constexpr int KSTEPS = K / 32;

  __shared__ __bf16 sA[2 * LBUF];
  __shared__ __bf16 sW[2 * LBUF];

  const int tid  = threadIdx.x;
  const int m0   = blockIdx.y * 128;
  const int n0   = blockIdx.x * 128;
  const int lane = tid & 31;
  const int wave = tid >> 5;
  const int wm   = (wave >> 2) * 64;   // 0 or 64
  const int wn   = (wave & 3) * 32;    // 0,32,64,96
  const int l15  = lane & 15;
  const int half = lane >> 4;

  // ---- staging: each thread owns rows (srow, srow+64), one 16B segment ----
  const int srow = tid >> 2;          // 0..63
  const int seg  = (tid & 3) * 8;     // bf16 elements
  const __bf16* gA0 = A + (size_t)(m0 + srow) * K + seg;
  const __bf16* gA1 = gA0 + (size_t)64 * K;
  const __bf16* gW0 = W + (size_t)(n0 + srow) * K + seg;
  const __bf16* gW1 = gW0 + (size_t)64 * K;
  const int sdst0 = srow * LDSW + seg;
  const int sdst1 = (srow + 64) * LDSW + seg;

  v8f acc[4][2] = {};

  // prologue: fetch tile 0 and park it in LDS buffer 0
  uint4 ra0 = *(const uint4*)gA0;
  uint4 ra1 = *(const uint4*)gA1;
  uint4 rw0 = *(const uint4*)gW0;
  uint4 rw1 = *(const uint4*)gW1;
  *(uint4*)(&sA[sdst0]) = ra0;
  *(uint4*)(&sA[sdst1]) = ra1;
  *(uint4*)(&sW[sdst0]) = rw0;
  *(uint4*)(&sW[sdst1]) = rw1;
  __syncthreads();

  int po = 0;  // ping-pong offset (elements)
  for (int ks = 0; ks < KSTEPS; ++ks) {
    // prefetch next K-tile into registers (overlaps with WMMA below)
    if (ks + 1 < KSTEPS) {
      const int off = (ks + 1) * 32;
      ra0 = *(const uint4*)(gA0 + off);
      ra1 = *(const uint4*)(gA1 + off);
      rw0 = *(const uint4*)(gW0 + off);
      rw1 = *(const uint4*)(gW1 + off);
    }

    // fragment loads + 8 WMMAs from current buffer
    v16bf af[4], bfr[2];
#pragma unroll
    for (int i = 0; i < 4; ++i) {
      const __bf16* r = &sA[po + (wm + i * 16 + l15) * LDSW];
      v8bf lo = *(const v8bf*)(r + 8 * half);        // K = 8*half + 0..7
      v8bf hi = *(const v8bf*)(r + 16 + 8 * half);   // K = 16 + 8*half + 0..7
      af[i] = make16(lo, hi);
    }
#pragma unroll
    for (int j = 0; j < 2; ++j) {
      const __bf16* r = &sW[po + (wn + j * 16 + l15) * LDSW];
      v8bf lo = *(const v8bf*)(r + 8 * half);        // K = 8*half + 0..7
      v8bf hi = *(const v8bf*)(r + 16 + 8 * half);   // K = 16 + 8*half + 0..7
      bfr[j] = make16(lo, hi);
    }
#pragma unroll
    for (int i = 0; i < 4; ++i)
#pragma unroll
      for (int j = 0; j < 2; ++j)
        acc[i][j] = __builtin_amdgcn_wmma_f32_16x16x32_bf16(
            false, af[i], false, bfr[j], (short)0, acc[i][j], false, false);

    // park next tile in the other buffer
    if (ks + 1 < KSTEPS) {
      const int qo = po ^ LBUF;
      *(uint4*)(&sA[qo + sdst0]) = ra0;
      *(uint4*)(&sA[qo + sdst1]) = ra1;
      *(uint4*)(&sW[qo + sdst0]) = rw0;
      *(uint4*)(&sW[qo + sdst1]) = rw1;
      __syncthreads();
      po = qo;
    }
  }

  // epilogue: D-layout lane L: n = l15, m = v + 8*half
#pragma unroll
  for (int j = 0; j < 2; ++j) {
    int n = n0 + wn + j * 16 + l15;
    float bv = bias[n];
#pragma unroll
    for (int i = 0; i < 4; ++i) {
#pragma unroll
      for (int v = 0; v < 8; ++v) {
        int m = m0 + wm + i * 16 + half * 8 + v;
        C[(size_t)m * N + n] = acc[i][j][v] + bv;
      }
    }
  }
}

// ---- GRU gate combine: h' = (1-z)*n + z*h ----------------------------------
// GI = x@W_ih^T + b_ih, GH = h@W_hh^T + b_hh (both [B, 3H], gates [r|z|n])

__global__ void k_combine(const float* __restrict__ GI, const float* __restrict__ GH,
                          float* __restrict__ hf, __bf16* __restrict__ hb) {
  int i = blockIdx.x * blockDim.x + threadIdx.x;  // < B_*HH
  int b = i >> 11;          // / HH
  int j = i & (HH - 1);
  size_t base = (size_t)b * H3 + j;
  float ir = GI[base],          hr = GH[base];
  float iz = GI[base + HH],     hz = GH[base + HH];
  float in_ = GI[base + 2*HH],  hn = GH[base + 2*HH];
  float r = sigmoidf_(ir + hr);
  float z = sigmoidf_(iz + hz);
  float n = tanhf(in_ + r * hn);
  float h = (1.0f - z) * n + z * hf[i];
  hf[i] = h;
  hb[i] = f2bf(h);
}

// ---- FC head: out[b,t,:] = sigmoid(h1 @ W_fc^T + b_fc); also next dec_in ----
// M=1024, K=2048, N=96. 8 blocks x 8 waves; wave: 16 rows x 6 n-tiles.
// Fragments loaded straight from global (W_fc ~0.4MB, L2-hot).

__global__ __launch_bounds__(256)
void k_fc(const __bf16* __restrict__ A, const __bf16* __restrict__ W,
          const float* __restrict__ bias, float* __restrict__ out,
          __bf16* __restrict__ xb, int t) {
  const int lane = threadIdx.x & 31;
  const int wave = threadIdx.x >> 5;
  const int l15 = lane & 15;
  const int half = lane >> 4;
  const int m0 = blockIdx.x * 128 + wave * 16;

  const __bf16* arow = A + (size_t)(m0 + l15) * HH;
  v8f acc[6] = {};

  for (int kb = 0; kb < HH; kb += 32) {
    v8bf alo = *(const v8bf*)(arow + kb + 8 * half);
    v8bf ahi = *(const v8bf*)(arow + kb + 16 + 8 * half);
    v16bf af = make16(alo, ahi);
#pragma unroll
    for (int j = 0; j < 6; ++j) {
      const __bf16* wrow = W + (size_t)(j * 16 + l15) * HH + kb;
      v8bf blo = *(const v8bf*)(wrow + 8 * half);
      v8bf bhi = *(const v8bf*)(wrow + 16 + 8 * half);
      v16bf bfg = make16(blo, bhi);
      acc[j] = __builtin_amdgcn_wmma_f32_16x16x32_bf16(
          false, af, false, bfg, (short)0, acc[j], false, false);
    }
  }

#pragma unroll
  for (int j = 0; j < 6; ++j) {
    int n = j * 16 + l15;
    float bv = bias[n];
#pragma unroll
    for (int v = 0; v < 8; ++v) {
      int m = m0 + half * 8 + v;
      float s = sigmoidf_(acc[j][v] + bv);
      out[(size_t)m * (T_ * OUT_) + t * OUT_ + n] = s;  // [B, T, OUT]
      xb[m * OUT_ + n] = f2bf(s);                       // next dec_in (detach)
    }
  }
}

// ---------------------------------------------------------------------------

extern "C" void kernel_launch(void* const* d_in, const int* in_sizes, int n_in,
                              void* d_out, int out_size, void* d_ws, size_t ws_size,
                              hipStream_t stream) {
  const float* input   = (const float*)d_in[0];
  // d_in[1] = future_poses (unused: teacher_forcing_rate == 0)
  const float* hiddens = (const float*)d_in[2];
  const float* W_ih0   = (const float*)d_in[3];
  const float* W_hh0   = (const float*)d_in[4];
  const float* b_ih0   = (const float*)d_in[5];
  const float* b_hh0   = (const float*)d_in[6];
  const float* W_ih1   = (const float*)d_in[7];
  const float* W_hh1   = (const float*)d_in[8];
  const float* b_ih1   = (const float*)d_in[9];
  const float* b_hh1   = (const float*)d_in[10];
  const float* W_fc    = (const float*)d_in[11];
  const float* b_fc    = (const float*)d_in[12];
  float* out = (float*)d_out;

  // workspace carve-out (256B aligned)
  char* ws = (char*)d_ws;
  auto carve = [&](size_t bytes) -> char* {
    char* p = ws;
    ws += (bytes + 255) & ~(size_t)255;
    return p;
  };
  __bf16* wih0 = (__bf16*)carve((size_t)H3 * IN_ * 2);
  __bf16* whh0 = (__bf16*)carve((size_t)H3 * HH * 2);
  __bf16* wih1 = (__bf16*)carve((size_t)H3 * HH * 2);
  __bf16* whh1 = (__bf16*)carve((size_t)H3 * HH * 2);
  __bf16* wfc  = (__bf16*)carve((size_t)OUT_ * HH * 2);
  float*  h0f  = (float*) carve((size_t)B_ * HH * 4);
  float*  h1f  = (float*) carve((size_t)B_ * HH * 4);
  __bf16* h0b  = (__bf16*)carve((size_t)B_ * HH * 2);
  __bf16* h1b  = (__bf16*)carve((size_t)B_ * HH * 2);
  __bf16* xb   = (__bf16*)carve((size_t)B_ * IN_ * 2);
  float*  GI   = (float*) carve((size_t)B_ * H3 * 4);
  float*  GH   = (float*) carve((size_t)B_ * H3 * 4);

  auto cvt = [&](const float* s, __bf16* d, int n) {
    k_cvt_bf16<<<(n + 255) / 256, 256, 0, stream>>>(s, d, n);
  };
  cvt(W_ih0, wih0, H3 * IN_);
  cvt(W_hh0, whh0, H3 * HH);
  cvt(W_ih1, wih1, H3 * HH);
  cvt(W_hh1, whh1, H3 * HH);
  cvt(W_fc,  wfc,  OUT_ * HH);
  cvt(input, xb,   B_ * IN_);
  k_init_h<<<(B_ * HH) / 256, 256, 0, stream>>>(hiddens, h0f, h1f, h0b, h1b);

  dim3 gBig(H3 / 128, B_ / 128);  // 48 x 8 = 384 workgroups per GEMM
  for (int t = 0; t < T_; ++t) {
    // layer 0
    k_gemm<IN_><<<gBig, 256, 0, stream>>>(xb,  wih0, b_ih0, GI);
    k_gemm<HH> <<<gBig, 256, 0, stream>>>(h0b, whh0, b_hh0, GH);
    k_combine<<<(B_ * HH) / 256, 256, 0, stream>>>(GI, GH, h0f, h0b);
    // layer 1
    k_gemm<HH> <<<gBig, 256, 0, stream>>>(h0b, wih1, b_ih1, GI);
    k_gemm<HH> <<<gBig, 256, 0, stream>>>(h1b, whh1, b_hh1, GH);
    k_combine<<<(B_ * HH) / 256, 256, 0, stream>>>(GI, GH, h1f, h1b);
    // output head + next dec_in
    k_fc<<<B_ / 128, 256, 0, stream>>>(h1b, wfc, b_fc, out, xb, t);
  }
}
